// ScaledDotProductAttention_41274635715223
// MI455X (gfx1250) — matrix operand order repaired and checked by reading, not compiled, not run on hardware
//
#include <hip/hip_runtime.h>

typedef __attribute__((ext_vector_type(16))) __bf16        v16bf;
typedef __attribute__((ext_vector_type(8)))  float         v8f;
typedef __attribute__((ext_vector_type(4)))  float         f32x4;
typedef __attribute__((ext_vector_type(4)))  unsigned int  u32x4;
typedef __attribute__((ext_vector_type(2)))  unsigned int  u32x2;

namespace {
constexpr int Lq  = 1024;   // sequence length
constexpr int Dk  = 512;    // head dim
constexpr int MT  = 32;     // query rows per workgroup
constexpr int KC  = 64;     // keys per streamed chunk
constexpr int NCH = Lq / KC;

// LDS layout (dynamic shared):
constexpr size_t OFF_Q   = 0;                              // 32*512 bf16  = 32768 B
constexpr size_t OFF_S   = OFF_Q  + (size_t)MT * Dk * 2;   // 32*1024 f32  = 131072 B
constexpr size_t OFF_P   = OFF_S  + (size_t)MT * Lq * 4;   // 32*1024 bf16 = 65536 B
constexpr size_t OFF_KV  = OFF_P  + (size_t)MT * Lq * 2;   // 64*512 bf16  = 65536 B
constexpr size_t OFF_RED = OFF_KV + (size_t)KC * Dk * 2;   // 32*8 f32     = 1024 B
constexpr size_t SMEM_BYTES = OFF_RED + (size_t)MT * 8 * 4;
}

__device__ __forceinline__ unsigned short f2bf(float f) {
  unsigned int u = __builtin_bit_cast(unsigned int, f);
  u += 0x7FFFu + ((u >> 16) & 1u);          // round-to-nearest-even
  return (unsigned short)(u >> 16);
}
__device__ __forceinline__ unsigned int pk2bf(float a, float b) {
  return (unsigned int)f2bf(a) | ((unsigned int)f2bf(b) << 16);
}

union V16U { v16bf v; u32x4 q[2]; };

// A operand: lane holds row (lane&15); halves [klo..klo+8) and [klo+16..klo+24)
__device__ __forceinline__ v16bf ldA(const unsigned short* p) {
  V16U t;
  t.q[0] = *(const u32x4*)(p);
  t.q[1] = *(const u32x4*)(p + 16);
  return t.v;
}
// B operand: lane holds column (lane&15); halves [klo..klo+8) and [klo+16..klo+24)
__device__ __forceinline__ v16bf ldB(const unsigned short* p) {
  V16U t;
  t.q[0] = *(const u32x4*)(p);
  t.q[1] = *(const u32x4*)(p + 16);
  return t.v;
}

__global__ void sdpa_wmma_bf16_kernel(const float* __restrict__ Q,
                                      const float* __restrict__ K,
                                      const float* __restrict__ V,
                                      const unsigned char* __restrict__ Msk,
                                      float* __restrict__ ctx_out,
                                      float* __restrict__ attn_out) {
  extern __shared__ char smem[];
  unsigned short* sQ  = (unsigned short*)(smem + OFF_Q);
  float*          sS  = (float*)(smem + OFF_S);
  unsigned short* sP  = (unsigned short*)(smem + OFF_P);
  unsigned short* sKV = (unsigned short*)(smem + OFF_KV);
  float*          red = (float*)(smem + OFF_RED);

  const int tid  = threadIdx.x;
  const int lane = tid & 31;
  const int wv   = tid >> 5;          // wave id 0..7 (wave32)
  const int hb   = lane >> 4;         // half-wave: 0/1
  const int ln16 = lane & 15;

  const int wg   = blockIdx.x;
  const int rb   = wg & (Lq / MT - 1);    // row block 0..31
  const int bh   = wg / (Lq / MT);        // fused (b,h) 0..31
  const int row0 = rb * MT;

  const float* Qg = Q + ((size_t)bh * Lq + row0) * Dk;
  const float* Kg = K + (size_t)bh * Lq * Dk;
  const float* Vg = V + (size_t)bh * Lq * Dk;
  const unsigned char* Mg = Msk + ((size_t)bh * Lq + row0) * (size_t)Lq;
  float* Cg = ctx_out  + ((size_t)bh * Lq + row0) * Dk;
  float* Ag = attn_out + ((size_t)bh * Lq + row0) * (size_t)Lq;

  const float scale = 0.044194173824159223f;  // 1/sqrt(512)

  // ---- load Q tile (32x512) as bf16, float4-vectorized ----
  for (int i = tid; i < MT * Dk / 4; i += 256) {
    f32x4 f = *(const f32x4*)(Qg + i * 4);
    u32x2 p; p.x = pk2bf(f[0], f[1]); p.y = pk2bf(f[2], f[3]);
    *(u32x2*)(sQ + i * 4) = p;
  }

  // ---- Stage A: S = scale * Q K^T, streamed over key chunks ----
  const int mrow = (wv & 1) * 16;       // wave's row tile within 32
  const int ncol = (wv >> 1) * 16;      // wave's col tile within chunk of 64
  for (int kc = 0; kc < NCH; ++kc) {
    __syncthreads();
    const float* kp = Kg + (size_t)(kc * KC) * Dk;
    for (int i = tid; i < KC * Dk / 4; i += 256) {
      f32x4 f = *(const f32x4*)(kp + i * 4);
      u32x2 p; p.x = pk2bf(f[0], f[1]); p.y = pk2bf(f[2], f[3]);
      *(u32x2*)(sKV + i * 4) = p;
    }
    __syncthreads();

    // prefetch next K chunk (one 128B line per prefetch)
    if (kc + 1 < NCH) {
      const float* kn = kp + (size_t)KC * Dk;
      for (int i = tid; i < KC * Dk / 32; i += 256)
        __builtin_prefetch(kn + i * 32, 0, 3);
    }

    v8f acc = {};
    const unsigned short* aRow = sQ  + (mrow + ln16) * Dk + hb * 8;
    const unsigned short* bRow = sKV + (ncol + ln16) * Dk + hb * 8;
#pragma unroll 4
    for (int d0 = 0; d0 < Dk; d0 += 32) {
      v16bf a = ldA(aRow + d0);
      v16bf b = ldB(bRow + d0);
      acc = __builtin_amdgcn_wmma_f32_16x16x32_bf16(false, a, false, b,
                                                    (short)0, acc, false, false);
    }
    const int col = kc * KC + ncol + ln16;
#pragma unroll
    for (int v = 0; v < 8; ++v) {
      const int m = mrow + v + hb * 8;                // C layout: VGPRv -> M=v / v+8
      sS[m * Lq + col] = acc[v] * scale;
    }
  }
  __syncthreads();

  // ---- Softmax over rows of sS (32 x 1024): 8 threads per row.
  //      Mask applied here (row-contiguous, vectorized) and written back. ----
  {
    const int r = tid >> 3, seg = tid & 7;
    float* row = sS + r * Lq;
    const unsigned char* mrw = Mg + (size_t)r * Lq;
    const int j0 = seg * 128, j1 = j0 + 128;

    float lmax = -3.4e38f;
    for (int j = j0; j < j1; j += 4) {
      const unsigned int m4 = *(const unsigned int*)(mrw + j);
      f32x4 s = *(f32x4*)(row + j);
#pragma unroll
      for (int t = 0; t < 4; ++t) {
        float v = ((m4 >> (8 * t)) & 0xFFu) ? -1e9f : s[t];  // mask true => -1e9
        s[t] = v;
        lmax = fmaxf(lmax, v);
      }
      *(f32x4*)(row + j) = s;
    }
    red[r * 8 + seg] = lmax;
    __syncthreads();
    float rmax = red[r * 8];
#pragma unroll
    for (int s = 1; s < 8; ++s) rmax = fmaxf(rmax, red[r * 8 + s]);
    __syncthreads();

    float lsum = 0.f;
    for (int j = j0; j < j1; j += 4) {
      f32x4 s = *(f32x4*)(row + j);
#pragma unroll
      for (int t = 0; t < 4; ++t) {
        s[t] = __expf(s[t] - rmax);
        lsum += s[t];
      }
      *(f32x4*)(row + j) = s;
    }
    red[r * 8 + seg] = lsum;
    __syncthreads();
    float rsum = 0.f;
#pragma unroll
    for (int s = 0; s < 8; ++s) rsum += red[r * 8 + s];
    const float inv = 1.0f / rsum;

    for (int j = j0; j < j1; j += 4) {
      f32x4 s = *(f32x4*)(row + j);
#pragma unroll
      for (int t = 0; t < 4; ++t) s[t] *= inv;
      *(f32x4*)(Ag + (size_t)r * Lq + j) = s;               // attn output
      u32x2 p; p.x = pk2bf(s[0], s[1]); p.y = pk2bf(s[2], s[3]);
      *(u32x2*)(sP + r * Lq + j) = p;                       // bf16 P for PV
    }
  }

  // ---- Stage B: context = P(32x1024) @ V(1024x512), V staged transposed ----
  v8f cacc[2][4] = {};
  const int d0w = wv * 64;                    // wave owns 64 output columns
  for (int kc = 0; kc < NCH; ++kc) {
    __syncthreads();
    const float* vp = Vg + (size_t)(kc * KC) * Dk;
    for (int i = tid; i < KC * Dk / 4; i += 256) {
      const int e = i * 4;
      const int k = e >> 9;                   // / Dk
      const int d = e & (Dk - 1);
      f32x4 f = *(const f32x4*)(vp + e);
#pragma unroll
      for (int t = 0; t < 4; ++t)
        sKV[(d + t) * KC + k] = f2bf(f[t]);   // Vt[d][k], row stride KC
    }
    __syncthreads();

    if (kc + 1 < NCH) {
      const float* vn = vp + (size_t)KC * Dk;
      for (int i = tid; i < KC * Dk / 32; i += 256)
        __builtin_prefetch(vn + i * 32, 0, 3);
    }

#pragma unroll
    for (int ks = 0; ks < KC; ks += 32) {
      const int kb = kc * KC + ks;
#pragma unroll
      for (int rt = 0; rt < 2; ++rt) {
        const unsigned short* aRow = sP + (rt * 16 + ln16) * Lq + kb + hb * 8;
#pragma unroll
        for (int dt = 0; dt < 4; ++dt) {
          const unsigned short* bRow = sKV + (d0w + dt * 16 + ln16) * KC + ks + hb * 8;
          v16bf a = ldA(aRow);
          v16bf b = ldB(bRow);
          cacc[rt][dt] = __builtin_amdgcn_wmma_f32_16x16x32_bf16(
              false, a, false, b, (short)0, cacc[rt][dt], false, false);
        }
      }
    }
  }

  // ---- write context (consecutive lanes -> coalesced b32 stores) ----
#pragma unroll
  for (int rt = 0; rt < 2; ++rt) {
#pragma unroll
    for (int dt = 0; dt < 4; ++dt) {
      const int dcol = d0w + dt * 16 + ln16;
#pragma unroll
      for (int v = 0; v < 8; ++v) {
        const int m = rt * 16 + v + hb * 8;
        Cg[(size_t)m * Dk + dcol] = cacc[rt][dt][v];
      }
    }
  }
}

extern "C" void kernel_launch(void* const* d_in, const int* in_sizes, int n_in,
                              void* d_out, int out_size, void* d_ws, size_t ws_size,
                              hipStream_t stream) {
  (void)in_sizes; (void)n_in; (void)out_size; (void)d_ws; (void)ws_size;
  const float*         Q = (const float*)d_in[0];
  const float*         K = (const float*)d_in[1];
  const float*         V = (const float*)d_in[2];
  const unsigned char* M = (const unsigned char*)d_in[3];

  constexpr int B = 4, H = 8;
  float* ctx  = (float*)d_out;                                   // [B,H,L,Dk]
  float* attn = (float*)d_out + (size_t)B * H * Lq * Dk;         // [B,H,L,L]

  hipFuncSetAttribute((const void*)sdpa_wmma_bf16_kernel,
                      hipFuncAttributeMaxDynamicSharedMemorySize,
                      (int)SMEM_BYTES);

  const int nblocks = B * H * (Lq / MT);     // 1024 workgroups
  sdpa_wmma_bf16_kernel<<<nblocks, 256, SMEM_BYTES, stream>>>(Q, K, V, M, ctx, attn);
}
